// Receptor_2946347565720
// MI455X (gfx1250) — matrix operand order repaired and checked by reading, not compiled, not run on hardware
//
#include <hip/hip_runtime.h>
#include <cstdint>
#include <cstddef>

// ---------------------------------------------------------------------------
// Shapes fixed by the reference: B=128, S=4096, I=128, H=256.
// x_freq is a runtime device scalar (chunk_len = x_freq / 25).
// ---------------------------------------------------------------------------
#define B_      128
#define S_      4096
#define I_      128
#define H_      256
#define FOURH   (4 * H_)
#define NT      (FOURH / 16)   // 64 N-tiles over the 4H gate dimension
#define KBI     (I_ / 32)      // 4  K-blocks for the input GEMM
#define KBH     (H_ / 32)      // 8  K-blocks for the recurrent GEMM
#define WAVES   (H_ / 16)      // 16 waves: wave w owns h columns [16w,16w+16)
#define BLOCK   (WAVES * 32)   // 512 threads (wave32)
#define GROUPS  (B_ / 16)      // 8 workgroups, one batch-group of 16 each

typedef __attribute__((ext_vector_type(16))) __bf16 v16bf;
typedef __attribute__((ext_vector_type(4)))  __bf16 v4bf;
typedef __attribute__((ext_vector_type(8)))  float  v8f;

#define LOG2E 1.4426950408889634f

// Branchless activations on the hardware transcendental pipe (no libm
// range-split branches on the serial recurrence path).
__device__ __forceinline__ float sigf(float x) {
    return __builtin_amdgcn_rcpf(1.0f + __builtin_amdgcn_exp2f(-x * LOG2E));
}
__device__ __forceinline__ float tanh_fast(float x) {
#if __has_builtin(__builtin_amdgcn_tanhf)
    return __builtin_amdgcn_tanhf(x);       // gfx1250 v_tanh_f32
#else
    // tanh(x) = 2*sigmoid(2x) - 1 ; saturates correctly at +-1, branchless
    return 2.0f * sigf(2.0f * x) - 1.0f;
#endif
}

// A-fragment-major LDS layout for a 16xK bf16 activation matrix:
//   element (row, k):  kb = k>>5, ko = k&31,
//   lane = row + 16*((ko>>3)&1),  slot j = (ko&7) + ((ko>>4)<<3)
//   half index = kb*512 + lane*16 + j
// so each lane's 16-half A fragment is one contiguous, 32B-aligned v16bf.

// ---------------------------------------------------------------------------
// Pack W_ih / W_hh (f32, [4H,K] row-major) into bf16 WMMA B-fragments,
// fragment-major: frag (ntile,kblk) -> 32 lanes x 16 halves contiguous.
// B-fragment lane layout (32x16 KxN tile): lane&15 = n, lane>>4 selects the
// K octets (0..7,16..23 / 8..15,24..31): k = 8*(lane>>4) + (j&7) + 16*(j>>3).
// Also fuses bias = b_ih + b_hh and zeroes the BN stats accumulator.
// ---------------------------------------------------------------------------
__global__ void pack_params(const float* __restrict__ Wih,
                            const float* __restrict__ Whh,
                            const float* __restrict__ bih,
                            const float* __restrict__ bhh,
                            __bf16* __restrict__ wihf,
                            __bf16* __restrict__ whhf,
                            float*  __restrict__ bias,
                            float*  __restrict__ stats) {
    int e = blockIdx.x * blockDim.x + threadIdx.x;
    if (e < NT * KBH * 512) {
        int f = e >> 9, r = e & 511, lane = r >> 4, j = r & 15;
        int ntile = f / KBH, kblk = f % KBH;
        int n = ntile * 16 + (lane & 15);
        int k = kblk * 32 + (j & 7) + ((lane >> 4) << 3) + ((j >> 3) << 4);
        whhf[e] = (__bf16)(Whh[(size_t)n * H_ + k]);     // W_hh^T[k][n]
    }
    if (e < NT * KBI * 512) {
        int f = e >> 9, r = e & 511, lane = r >> 4, j = r & 15;
        int ntile = f / KBI, kblk = f % KBI;
        int n = ntile * 16 + (lane & 15);
        int k = kblk * 32 + (j & 7) + ((lane >> 4) << 3) + ((j >> 3) << 4);
        wihf[e] = (__bf16)(Wih[(size_t)n * I_ + k]);     // W_ih^T[k][n]
    }
    if (e < FOURH) bias[e] = bih[e] + bhh[e];
    if (e < 2 * H_) stats[e] = 0.0f;
}

// ---------------------------------------------------------------------------
// Persistent recurrent kernel: one workgroup per 16-batch group.
// M=16 WMMA tile = batch group; wave w owns h cols [16w,16w+16) and hence the
// matching i/f/g/o gate tiles -> the entire cell update is register-local.
// h state: bf16 in LDS, A-fragment-major and DOUBLE-BUFFERED so one barrier
// per step suffices (step-t reads and step-t writes hit different buffers).
// x_t: cooperatively staged f32->bf16 into double-buffered LDS (each tile is
// loaded from HBM exactly once per workgroup, not once per wave).
// c state: fp32 in VGPRs (C/D tile layout). 48 v_wmma_bf16 per wave per step.
// ---------------------------------------------------------------------------
__global__ void __launch_bounds__(BLOCK)
lstm_wmma(const float* __restrict__ x,
          const __bf16* __restrict__ wihf,
          const __bf16* __restrict__ whhf,
          const float*  __restrict__ bias,
          const int*    __restrict__ xfreq,
          float* __restrict__ out) {
    __shared__ __bf16 h_lds[2][KBH * 512];     // 16 KB: h state, frag-major
    __shared__ __bf16 x_lds[2][KBI * 512];     //  8 KB: x_t staging

    const int tid  = threadIdx.x;
    const int lane = tid & 31;
    const int w    = tid >> 5;       // wave id: h columns [16w, 16w+16)
    const int m    = lane & 15;      // C-tile N index (and frag column)
    const int hi   = lane >> 4;      // half-wave selector (C-tile M offset 8)
    const int b0   = blockIdx.x * 16;

    const int chunk_len = (*xfreq) / 25;     // RECEPTOR_FREQ = 25
    const int n_chunks  = S_ / chunk_len;

    // --- x staging role for this thread: 4 consecutive k of one row ---
    const int srow = tid >> 5;                  // 0..15
    const int sk   = (tid & 31) * 4;            // 0..124
    const int skb  = sk >> 5;
    const int sko  = sk & 31;
    const int shalf = (sko >> 3) & 1;
    const int sj   = (sko & 7) + ((sko >> 4) << 3);
    const int sdst = skb * 512 + (srow + (shalf << 4)) * 16 + sj;
    const float* sxrow = x + (size_t)(b0 + srow) * S_ * I_ + sk;

    // --- h write-back mapping for this lane: fixed k = 16w + m ---
    const int hk   = w * 16 + m;
    const int hkb  = hk >> 5;
    const int hko  = hk & 31;
    const int hhalf = (hko >> 3) & 1;
    const int hj   = (hko & 7) + ((hko >> 4) << 3);
    const int hbase = hkb * 512 + (hhalf << 8) + hj;   // + row*16 per register

    // zero h state buffer 0, stage x_0 into buffer 0
    for (int i = tid; i < KBH * 512; i += BLOCK) h_lds[0][i] = (__bf16)0.0f;
    {
        float4 v = *(const float4*)(sxrow);
        v4bf b; b[0] = (__bf16)v.x; b[1] = (__bf16)v.y;
                b[2] = (__bf16)v.z; b[3] = (__bf16)v.w;
        *(v4bf*)&x_lds[0][sdst] = b;
    }
    __syncthreads();

    float bias_g[4];
#pragma unroll
    for (int g = 0; g < 4; ++g)
        bias_g[g] = bias[g * H_ + w * 16 + m];   // per-column gate bias

    // per-gate weight fragment pointers (lane folded in)
    const v16bf* wp_i[4];
    const v16bf* wp_h[4];
#pragma unroll
    for (int g = 0; g < 4; ++g) {
        const int ntile = g * (H_ / 16) + w;
        wp_i[g] = (const v16bf*)wihf + (size_t)(ntile * KBI) * 32 + lane;
        wp_h[g] = (const v16bf*)whhf + (size_t)(ntile * KBH) * 32 + lane;
    }

    v8f cst = {};                                 // c state, fp32, C layout

    int tc = 0, chunk = 0;
#pragma unroll 1
    for (int t = 0; t < S_; ++t) {
        const int cur = t & 1;

        // ---- stage x_{t+1} into the other buffer (overlaps the GEMM) ----
        if (t + 1 < S_) {
            const float* p = sxrow + (size_t)(t + 1) * I_;
            if (t + 3 < S_)   // gfx1250 global_prefetch_b8 two steps ahead
                __builtin_prefetch((const char*)(sxrow + (size_t)(t + 3) * I_), 0, 0);
            float4 v = *(const float4*)(p);
            v4bf b; b[0] = (__bf16)v.x; b[1] = (__bf16)v.y;
                    b[2] = (__bf16)v.z; b[3] = (__bf16)v.w;
            *(v4bf*)&x_lds[cur ^ 1][sdst] = b;
        }

        // ---- init accumulators with bias ----
        v8f acc[4];
#pragma unroll
        for (int g = 0; g < 4; ++g) {
            v8f a;
#pragma unroll
            for (int r = 0; r < 8; ++r) a[r] = bias_g[g];
            acc[g] = a;
        }

        // ---- input-to-hidden: gates += x_t @ W_ih^T ----
#pragma unroll 2
        for (int kb = 0; kb < KBI; ++kb) {
            const v16bf ax = *(const v16bf*)&x_lds[cur][kb * 512 + lane * 16];
#pragma unroll
            for (int g = 0; g < 4; ++g) {
                const v16bf bf = wp_i[g][kb * 32];
                acc[g] = __builtin_amdgcn_wmma_f32_16x16x32_bf16(
                    false, ax, false, bf, (short)0, acc[g], false, false);
            }
        }

        // ---- hidden-to-hidden: gates += h @ W_hh^T ----
#pragma unroll 2
        for (int kb = 0; kb < KBH; ++kb) {
            const v16bf ah = *(const v16bf*)&h_lds[cur][kb * 512 + lane * 16];
#pragma unroll
            for (int g = 0; g < 4; ++g) {
                const v16bf bf = wp_h[g][kb * 32];
                acc[g] = __builtin_amdgcn_wmma_f32_16x16x32_bf16(
                    false, ah, false, bf, (short)0, acc[g], false, false);
            }
        }

        // ---- cell update; C/D layout: M = r + 8*hi, N = m ----
        // writes go to h buffer cur^1: no conflict with this step's reads,
        // so a single end-of-step barrier publishes h_{t+1} and the x stage.
        const bool boundary = (tc == chunk_len - 1);
        const float carry   = boundary ? 0.5f : 1.0f;   // RESIDUAL
#pragma unroll
        for (int r = 0; r < 8; ++r) {
            float ig = sigf(acc[0][r]);
            float fg = sigf(acc[1][r]);
            float gg = tanh_fast(acc[2][r]);
            float og = sigf(acc[3][r]);
            float cn = fg * cst[r] + ig * gg;
            float hn = og * tanh_fast(cn);
            cst[r] = cn * carry;
            const int M = r + hi * 8;
            h_lds[cur ^ 1][hbase + M * 16] = (__bf16)(hn * carry);
            if (boundary)
                out[((size_t)(b0 + M) * n_chunks + chunk) * H_ + w * 16 + m] = hn;
        }
        __syncthreads();   // h_{t+1} and x_{t+1} published

        if (boundary) { tc = 0; ++chunk; } else { ++tc; }
    }
}

// ---------------------------------------------------------------------------
// BatchNorm over flattened [R, H]: per-channel sum / sumsq, then normalize.
// ---------------------------------------------------------------------------
__global__ void bn_reduce(const float* __restrict__ o,
                          float* __restrict__ stats, int R) {
    const int h = threadIdx.x;   // blockDim.x == H_
    float s = 0.0f, s2 = 0.0f;
    for (int r = blockIdx.x; r < R; r += gridDim.x) {
        float v = o[(size_t)r * H_ + h];
        s += v; s2 += v * v;
    }
    atomicAdd(&stats[h], s);
    atomicAdd(&stats[H_ + h], s2);
}

__global__ void bn_norm(float* __restrict__ o,
                        const float* __restrict__ stats,
                        const float* __restrict__ gamma,
                        const float* __restrict__ beta, int R) {
    size_t idx = (size_t)blockIdx.x * blockDim.x + threadIdx.x;
    size_t total = (size_t)R * H_;
    if (idx >= total) return;
    int h = (int)(idx & (H_ - 1));
    float invR = 1.0f / (float)R;
    float mean = stats[h] * invR;
    float var  = stats[H_ + h] * invR - mean * mean;
    o[idx] = (o[idx] - mean) * rsqrtf(var + 1e-5f) * gamma[h] + beta[h];
}

// ---------------------------------------------------------------------------
extern "C" void kernel_launch(void* const* d_in, const int* in_sizes, int n_in,
                              void* d_out, int out_size, void* d_ws, size_t ws_size,
                              hipStream_t stream) {
    (void)in_sizes; (void)n_in; (void)ws_size;

    const float* x     = (const float*)d_in[0];
    const float* Wih   = (const float*)d_in[1];
    const float* Whh   = (const float*)d_in[2];
    const float* bih   = (const float*)d_in[3];
    const float* bhh   = (const float*)d_in[4];
    const float* gamma = (const float*)d_in[5];
    const float* beta  = (const float*)d_in[6];
    const int*   xfreq = (const int*)d_in[7];
    float* out = (float*)d_out;

    // workspace layout (< 1 MB): bf16 weight fragments + fused bias + BN stats
    __bf16* wihf = (__bf16*)d_ws;
    __bf16* whhf = wihf + (size_t)NT * KBI * 512;
    float*  bias = (float*)(whhf + (size_t)NT * KBH * 512);
    float*  stats = bias + FOURH;

    const int packTotal = NT * KBH * 512;   // largest pack job (262144)
    pack_params<<<packTotal / 256, 256, 0, stream>>>(
        Wih, Whh, bih, bhh, wihf, whhf, bias, stats);

    lstm_wmma<<<GROUPS, BLOCK, 0, stream>>>(x, wihf, whhf, bias, xfreq, out);

    const int R = out_size / H_;            // B * n_chunks
    bn_reduce<<<2048, H_, 0, stream>>>(out, stats, R);
    bn_norm<<<(out_size + 255) / 256, 256, 0, stream>>>(out, stats, gamma, beta, R);
}
